// SimpleRNN_661424964144
// MI455X (gfx1250) — matrix operand order repaired and checked by reading, not compiled, not run on hardware
//
#include <hip/hip_runtime.h>
#include <hip/hip_bf16.h>

typedef __attribute__((ext_vector_type(16))) __bf16 v16bf;
typedef __attribute__((ext_vector_type(8)))  __bf16 v8bf;
typedef __attribute__((ext_vector_type(8)))  float  v8f;

#define VECTN  77
#define HID    1024
#define TSTEPS 512
#define BATCH  128
#define NG     4096      // 4*HID gate columns
#define KX     96        // padded input K (77 -> 96)
#define NKT    35        // (96+1024)/32 K-tiles
#define NTILES 256       // NG/16 N-tiles

static __device__ __forceinline__ float sigmoidf_(float x) {
    return 1.0f / (1.0f + __expf(-x));
}
static __device__ __forceinline__ float tanhf_(float x) {
    float e = __expf(2.0f * x);
    return (e - 1.0f) / (e + 1.0f);
}

// ---------------- prep kernels ----------------

__global__ void k_zero(unsigned int* p, int n) {
    int i = blockIdx.x * blockDim.x + threadIdx.x;
    if (i < n) p[i] = 0u;
}

// Pack A = inp_w.T (96x1024, zero-padded rows >=77) into WMMA A-fragment layout.
__global__ void k_pack_awc(const float* __restrict__ inp_w, __bf16* __restrict__ awc) {
    int idx = blockIdx.x * blockDim.x + threadIdx.x;
    if (idx >= 32 * 6 * 512) return;
    int kt = idx / (6 * 512);
    int rem = idx % (6 * 512);
    int mt = rem / 512;
    int s  = rem % 512;
    int l = s >> 4, i = s & 15;
    int m = mt * 16 + (l & 15);
    int k = kt * 32 + ((l >= 16) ? 8 : 0) + (i & 7) + ((i >= 8) ? 16 : 0);
    float v = (m < VECTN) ? inp_w[k * VECTN + m] : 0.0f;
    awc[idx] = (__bf16)v;
}

// Wc[96,4096] = inp_w.T @ w_ih.T via WMMA (B frags built on the fly from w_ih rows).
__global__ void k_wc_gemm(const __bf16* __restrict__ awc, const float* __restrict__ w_ih,
                          float* __restrict__ wc) {
    int c = blockIdx.x;                    // 64 blocks, 64 output cols each
    int tid = threadIdx.x;                 // 192 threads = 6 waves
    int w = tid >> 5, l = tid & 31;
    int mt = w;
    v8f acc[4] = {};
    for (int kt = 0; kt < 32; ++kt) {
        v16bf a = *(const v16bf*)(awc + (((size_t)kt * 6 + mt) * 512 + l * 16));
        int kbase = kt * 32 + ((l >= 16) ? 8 : 0);
#pragma unroll
        for (int nt = 0; nt < 4; ++nt) {
            int j = c * 64 + nt * 16 + (l & 15);
            const float* wr = w_ih + (size_t)j * HID + kbase;
            v16bf b;
#pragma unroll
            for (int q = 0; q < 16; ++q) b[q] = (__bf16)wr[(q & 7) + ((q >= 8) ? 16 : 0)];
            acc[nt] = __builtin_amdgcn_wmma_f32_16x16x32_bf16(
                false, a, false, b, (short)0, acc[nt], false, false);
        }
    }
#pragma unroll
    for (int nt = 0; nt < 4; ++nt) {
        int n = c * 64 + nt * 16 + (l & 15);
#pragma unroll
        for (int r = 0; r < 8; ++r) {
            int m = mt * 16 + r + ((l >= 16) ? 8 : 0);
            wc[m * NG + n] = acc[nt][r];
        }
    }
}

// Combined bias, permuted: bc[n'] with n' = 4*hcol + gate, j = gate*1024 + hcol
__global__ void k_bias(const float* __restrict__ inp_b, const float* __restrict__ w_ih,
                       const float* __restrict__ b_ih, const float* __restrict__ b_hh,
                       float* __restrict__ bc) {
    int np = blockIdx.x * blockDim.x + threadIdx.x;
    if (np >= NG) return;
    int j = (np & 3) * HID + (np >> 2);
    float acc = b_ih[j] + b_hh[j];
    const float* wr = w_ih + (size_t)j * HID;
    for (int h = 0; h < HID; ++h) acc += inp_b[h] * wr[h];
    bc[np] = acc;
}

// Pack combined gate weight [1120 x 4096] (rows<96: Wc, rows>=96: w_hh) into
// B-fragment layout, with gate-interleaved column permutation n' = 4*hcol + gate.
__global__ void k_bpack(const float* __restrict__ wc, const float* __restrict__ w_hh,
                        __bf16* __restrict__ bp) {
    int idx = blockIdx.x * blockDim.x + threadIdx.x;
    if (idx >= NKT * NTILES * 512) return;
    int kt = idx / (NTILES * 512);
    int rem = idx % (NTILES * 512);
    int ntile = rem / 512;
    int s = rem % 512;
    int l = s >> 4, i = s & 15;
    int k = kt * 32 + ((l >= 16) ? 8 : 0) + (i & 7) + ((i >= 8) ? 16 : 0);
    int np = ntile * 16 + (l & 15);
    int j = (np & 3) * HID + (np >> 2);
    float v;
    if (k < KX) v = wc[k * NG + j];
    else        v = w_hh[(size_t)j * HID + (k - KX)];
    bp[idx] = (__bf16)v;
}

// Pack out_w.T (1024 x 80, cols>=77 zero) into B-fragment layout.
__global__ void k_outwp(const float* __restrict__ out_w, __bf16* __restrict__ op) {
    int idx = blockIdx.x * blockDim.x + threadIdx.x;
    if (idx >= 32 * 5 * 512) return;
    int kt = idx / (5 * 512);
    int rem = idx % (5 * 512);
    int nt = rem / 512;
    int s = rem % 512;
    int l = s >> 4, i = s & 15;
    int k = kt * 32 + ((l >= 16) ? 8 : 0) + (i & 7) + ((i >= 8) ? 16 : 0);
    int v = nt * 16 + (l & 15);
    float val = (v < VECTN) ? out_w[v * HID + k] : 0.0f;
    op[idx] = (__bf16)val;
}

// Pack inputs for ALL timesteps into A-fragment layout (K tiles 0..2, K padded to 96).
__global__ void k_xpack(const float* __restrict__ inputs, __bf16* __restrict__ xp) {
    int idx = blockIdx.x * blockDim.x + threadIdx.x;
    if (idx >= TSTEPS * 3 * 8 * 512) return;
    int t = idx / (3 * 8 * 512);
    int rem = idx % (3 * 8 * 512);
    int kt = rem / (8 * 512);
    int rem2 = rem % (8 * 512);
    int mt = rem2 / 512;
    int s = rem2 % 512;
    int l = s >> 4, i = s & 15;
    int m = mt * 16 + (l & 15);
    int k = kt * 32 + ((l >= 16) ? 8 : 0) + (i & 7) + ((i >= 8) ? 16 : 0);
    float v = (k < VECTN) ? inputs[((size_t)t * BATCH + m) * VECTN + k] : 0.0f;
    xp[idx] = (__bf16)v;
}

// ---------------- per-timestep fused kernel ----------------
// Blocks 0..63 : gate GEMM (64 n'-cols = 16 hidden cols x 4 gates) + LSTM pointwise
//                + scatter new hx into next step's A-fragment buffer.
// Block 64     : output projection of the PREVIOUS step's hx (concurrent).
__global__ __launch_bounds__(256)
void k_step(int t,
            const __bf16* __restrict__ bp,   // gate B-pack
            const __bf16* __restrict__ xp,   // x A-pack, all t
            const __bf16* __restrict__ hin,  // hx A-pack (current)
            __bf16*       __restrict__ hout, // hx A-pack (next)
            float*        __restrict__ cx,
            const float*  __restrict__ bc,
            const __bf16* __restrict__ owp,  // out_w B-pack
            const float*  __restrict__ out_b,
            float*        __restrict__ out) {
    __shared__ __bf16 bstage[4 * 512];   // 4 B-tiles (32x16 each) = 4 KB
    __shared__ float  Cl[128 * 64];      // C dump for pointwise = 32 KB
    int tid = threadIdx.x;
    int l = tid & 31;
    int w = tid >> 5;

    if (blockIdx.x < 64) {
        if (t >= TSTEPS) return;
        int c = blockIdx.x;
        int mt = w;                      // 8 waves -> 8 M-tiles (batch = 128)

        // B staging: all 256 threads move 16 B each (1x b128 + 1x ds_store_b128)
        int sf = tid >> 6;               // fragment 0..3
        int sl = (tid >> 1) & 31;        // lane slot
        int sh = tid & 1;                // half (8 bf16)
        size_t soff = ((size_t)(c * 4 + sf) * 32 + sl) * 16 + sh * 8;
        int dstoff = (sf * 32 + sl) * 16 + sh * 8;

        // software pipeline: registers hold tile kt+1 while WMMA consumes tile kt
        v8bf breg = *(const v8bf*)(bp + (size_t)0 * NTILES * 512 + soff);
        const __bf16* a0 = xp + (((size_t)t * 3 + 0) * 8 + mt) * 512 + l * 16;
        v16bf areg = *(const v16bf*)a0;
        v16bf anext = areg;

        v8f acc[4] = {};
        for (int kt = 0; kt < NKT; ++kt) {
            __syncthreads();             // previous tile's readers done
            *(v8bf*)&bstage[dstoff] = breg;
            __syncthreads();
            if (kt + 1 < NKT) {          // issue next tile's loads; latency hides
                breg = *(const v8bf*)(bp + (size_t)(kt + 1) * NTILES * 512 + soff);
                const __bf16* asrc = (kt + 1 < 3)
                    ? xp + (((size_t)t * 3 + (kt + 1)) * 8 + mt) * 512 + l * 16
                    : hin + ((((size_t)(kt + 1 - 3)) * 8 + mt) * 512 + l * 16);
                anext = *(const v16bf*)asrc;
            }
#pragma unroll
            for (int nt = 0; nt < 4; ++nt) {
                v16bf b = *(const v16bf*)&bstage[(nt * 32 + l) * 16];
                acc[nt] = __builtin_amdgcn_wmma_f32_16x16x32_bf16(
                    false, areg, false, b, (short)0, acc[nt], false, false);
            }
            areg = anext;
        }
        __syncthreads();
#pragma unroll
        for (int nt = 0; nt < 4; ++nt) {
#pragma unroll
            for (int r = 0; r < 8; ++r) {
                int m = mt * 16 + r + ((l >= 16) ? 8 : 0);
                int nl = nt * 16 + (l & 15);
                Cl[m * 64 + nl] = acc[nt][r];
            }
        }
        __syncthreads();
        // LSTM pointwise: 128 rows x 16 hidden cols, 8 cells / thread
#pragma unroll
        for (int q = 0; q < 8; ++q) {
            int e = q * 256 + tid;
            int m = e >> 4;
            int hl = e & 15;
            int nb = c * 64 + hl * 4;
            float iv = sigmoidf_(Cl[m * 64 + hl * 4 + 0] + bc[nb + 0]);
            float fv = sigmoidf_(Cl[m * 64 + hl * 4 + 1] + bc[nb + 1]);
            float gv = tanhf_   (Cl[m * 64 + hl * 4 + 2] + bc[nb + 2]);
            float ov = sigmoidf_(Cl[m * 64 + hl * 4 + 3] + bc[nb + 3]);
            int hcol = c * 16 + hl;
            int ci = m * HID + hcol;
            float cnew = fv * cx[ci] + iv * gv;
            float hnew = ov * tanhf_(cnew);
            cx[ci] = cnew;
            // scatter into next-step A-fragment layout (k = 96 + hcol)
            int kl = hcol & 31;
            int la = (m & 15) + ((kl & 8) ? 16 : 0);
            int ia = (kl & 7) | ((kl & 16) >> 1);
            hout[(((size_t)(hcol >> 5) * 8 + (m >> 4)) * 512 + la * 16 + ia)] = (__bf16)hnew;
            if (t == TSTEPS - 1) {
                out[(size_t)TSTEPS * BATCH * VECTN + ci] = hnew;                    // hx
                out[(size_t)TSTEPS * BATCH * VECTN + BATCH * HID + ci] = cnew;      // cx
            }
        }
    } else {
        // output projection: out[t-1] = hx_t @ out_w.T + out_b  (hx_t lives in hin)
        if (t < 1) return;
        int mt = w;
        v8f acc[5] = {};
        v16bf a = *(const v16bf*)(hin + (((size_t)0 * 8 + mt) * 512 + l * 16));
        for (int kt = 0; kt < 32; ++kt) {
            v16bf anext = a;
            if (kt + 1 < 32) {
                anext = *(const v16bf*)(hin + (((size_t)(kt + 1) * 8 + mt) * 512 + l * 16));
                __builtin_prefetch(owp + (((size_t)(kt + 1) * 5) * 32 + l) * 16, 0, 0);
            }
#pragma unroll
            for (int nt = 0; nt < 5; ++nt) {
                v16bf b = *(const v16bf*)(owp + (((size_t)kt * 5 + nt) * 32 + l) * 16);
                acc[nt] = __builtin_amdgcn_wmma_f32_16x16x32_bf16(
                    false, a, false, b, (short)0, acc[nt], false, false);
            }
            a = anext;
        }
        int tp = t - 1;
#pragma unroll
        for (int nt = 0; nt < 5; ++nt) {
            int n = nt * 16 + (l & 15);
            if (n < VECTN) {
                float ob = out_b[n];
#pragma unroll
                for (int r = 0; r < 8; ++r) {
                    int m = mt * 16 + r + ((l >= 16) ? 8 : 0);
                    out[((size_t)tp * BATCH + m) * VECTN + n] = acc[nt][r] + ob;
                }
            }
        }
    }
}

extern "C" void kernel_launch(void* const* d_in, const int* in_sizes, int n_in,
                              void* d_out, int out_size, void* d_ws, size_t ws_size,
                              hipStream_t stream) {
    const float* inputs = (const float*)d_in[0];
    const float* inp_w  = (const float*)d_in[1];
    const float* inp_b  = (const float*)d_in[2];
    const float* w_ih   = (const float*)d_in[3];
    const float* b_ih   = (const float*)d_in[4];
    const float* w_hh   = (const float*)d_in[5];
    const float* b_hh   = (const float*)d_in[6];
    const float* out_w  = (const float*)d_in[7];
    const float* out_b  = (const float*)d_in[8];
    float* out = (float*)d_out;
    (void)in_sizes; (void)n_in; (void)out_size; (void)ws_size;

    char* ws = (char*)d_ws;
    size_t o = 0;
    auto take = [&](size_t bytes) -> char* {
        char* p = ws + o;
        o = (o + bytes + 255) & ~(size_t)255;
        return p;
    };
    __bf16* bp  = (__bf16*)take((size_t)NKT * NTILES * 512 * 2);   // ~8.75 MB
    __bf16* owp = (__bf16*)take((size_t)32 * 5 * 512 * 2);         // 160 KB
    __bf16* xp  = (__bf16*)take((size_t)TSTEPS * 3 * 8 * 512 * 2); // 12 MB
    __bf16* hx0 = (__bf16*)take((size_t)32 * 8 * 512 * 2);         // 256 KB
    __bf16* hx1 = (__bf16*)take((size_t)32 * 8 * 512 * 2);         // 256 KB
    float*  cx  = (float*) take((size_t)BATCH * HID * 4);          // 512 KB
    float*  wc  = (float*) take((size_t)KX * NG * 4);              // 1.5 MB
    float*  bc  = (float*) take((size_t)NG * 4);
    __bf16* awc = (__bf16*)take((size_t)32 * 6 * 512 * 2);

    // hx0, hx1, cx are contiguous: zero them all (hx_0 = cx_0 = 0)
    int zero_words = (32 * 8 * 512 * 2 * 2 + BATCH * HID * 4) / 4;
    k_zero<<<(zero_words + 255) / 256, 256, 0, stream>>>((unsigned int*)hx0, zero_words);

    k_pack_awc<<<(32 * 6 * 512 + 255) / 256, 256, 0, stream>>>(inp_w, awc);
    k_wc_gemm<<<64, 192, 0, stream>>>(awc, w_ih, wc);
    k_bias<<<(NG + 255) / 256, 256, 0, stream>>>(inp_b, w_ih, b_ih, b_hh, bc);
    k_bpack<<<(NKT * NTILES * 512 + 255) / 256, 256, 0, stream>>>(wc, w_hh, bp);
    k_outwp<<<(32 * 5 * 512 + 255) / 256, 256, 0, stream>>>(out_w, owp);
    k_xpack<<<(TSTEPS * 3 * 8 * 512 + 255) / 256, 256, 0, stream>>>(inputs, xp);

    for (int t = 0; t <= TSTEPS; ++t) {
        const __bf16* hin = (t & 1) ? hx1 : hx0;
        __bf16*       hout = (t & 1) ? hx0 : hx1;
        k_step<<<65, 256, 0, stream>>>(t, bp, xp, hin, hout, cx, bc, owp, out_b, out);
    }
}
